// MultiHeadAttention_68779606278690
// MI455X (gfx1250) — matrix operand order repaired and checked by reading, not compiled, not run on hardware
//
#include <hip/hip_runtime.h>
#include <hip/hip_bf16.h>
#include <stdint.h>

// Problem constants (fixed by the reference).
#define BB   2
#define SS   2048
#define HQn  16
#define HKVn 8
#define HDn  256
#define DDn  4096   /* HQ*HD */
#define DKVn 2048   /* HKV*HD */
#define BSn  4096   /* B*S    */

typedef __bf16 bf16x8  __attribute__((ext_vector_type(8)));
typedef __bf16 bf16x16 __attribute__((ext_vector_type(16)));
typedef float  f32x8   __attribute__((ext_vector_type(8)));

// bf16 <-> f32 via bit ops only (RNE), no native bf16 arithmetic needed.
__device__ inline __bf16 f2bf(float f) {
  union { float f; uint32_t u; } v; v.f = f;
  uint32_t r = v.u + 0x7FFFu + ((v.u >> 16) & 1u);
  uint16_t h = (uint16_t)(r >> 16);
  return __builtin_bit_cast(__bf16, h);
}

__device__ inline bf16x16 cat8(bf16x8 lo, bf16x8 hi) {
  return __builtin_shufflevector(lo, hi, 0,1,2,3,4,5,6,7,8,9,10,11,12,13,14,15);
}

__device__ inline f32x8 wmma_bf16(bf16x16 a, bf16x16 b, f32x8 c) {
  // (neg_a, A, neg_b, B, c_mod, C, reuse_a, reuse_b)
  return __builtin_amdgcn_wmma_f32_16x16x32_bf16(false, a, false, b, (short)0, c,
                                                 false, false);
}

// Async copy of one 16B chunk global->LDS (ASYNCcnt-tracked, no VGPR data).
__device__ inline void async_cp16(const __bf16* g, const __bf16* l) {
  uint32_t loff = (uint32_t)(uintptr_t)l;   // low 32 bits of generic = LDS offset
  uint64_t gptr = (uint64_t)(uintptr_t)g;
  asm volatile("global_load_async_to_lds_b128 %0, %1, off"
               :: "v"(loff), "v"(gptr) : "memory");
}

// ---------------------------------------------------------------- converts
__global__ __launch_bounds__(256) void k_cvt_bf16(const float* __restrict__ in,
                                                  __bf16* __restrict__ out, int n) {
  int i = blockIdx.x * 256 + threadIdx.x;
  if (i < n) out[i] = f2bf(in[i]);
}

// Wt[n*K + k] = bf16(W[k*N + n])  (writes coalesced)
__global__ __launch_bounds__(256) void k_transpose_bf16(const float* __restrict__ W,
                                                        __bf16* __restrict__ Wt,
                                                        int N, int K) {
  int i = blockIdx.x * 256 + threadIdx.x;
  if (i < N * K) {
    int n = i / K, k = i - n * K;
    Wt[i] = f2bf(W[(size_t)k * N + n]);
  }
}

// ---------------------------------------------------------------- WMMA GEMM
// C[M,N] = A[M,K] * Bt[N,K]^T.  256 thr = 8 waves; block tile 128x128;
// wave tile 64x32 (4x2 of 16x16) -> 8 v_wmma per 32-wide K step.
// A/B tiles double-buffered in LDS via global_load_async_to_lds_b128;
// next K-step copy overlaps current WMMAs (split-phase via s_wait_asynccnt).
__global__ __launch_bounds__(256) void k_gemm_bf16(const __bf16* __restrict__ A,
                                                   const __bf16* __restrict__ Bt,
                                                   float* __restrict__ C,
                                                   int M, int N, int K) {
  __shared__ __align__(16) __bf16 tA[2][128 * 32];  // 2 x 8KB
  __shared__ __align__(16) __bf16 tB[2][128 * 32];  // 2 x 8KB

  int tid  = threadIdx.x;
  int lane = tid & 31;
  int wave = tid >> 5;
  int wm = wave >> 2, wn = wave & 3;
  int rowBase = blockIdx.y * 128;
  int colBase = blockIdx.x * 128;
  const int arow = lane & 15;
  const int akh  = (lane < 16) ? 0 : 8;   // A-frag K-half (ISA 16-bit A layout)
  const int bkh  = (lane < 16) ? 0 : 8;   // B-frag K-half (ISA 16-bit B layout)
  const __bf16* Ab = A  + (size_t)rowBase * K;
  const __bf16* Bb = Bt + (size_t)colBase * K;

  // Stage one 128x32 A tile + 128x32 B tile into LDS buffer `buf`.
  // 256 threads x 2 iters x (1 A + 1 B) = 4 async ops per thread per stage.
  auto stage = [&](int buf, int k0) {
#pragma unroll
    for (int i = 0; i < 2; i++) {
      int idx = i * 256 + tid;  // 0..511
      int row = idx >> 2;       // 0..127
      int ch  = idx & 3;        // 16B chunk within 64B row
      async_cp16(Ab + (size_t)row * K + k0 + ch * 8, &tA[buf][row * 32 + ch * 8]);
      async_cp16(Bb + (size_t)row * K + k0 + ch * 8, &tB[buf][row * 32 + ch * 8]);
    }
  };

  const f32x8 zf = {};
  f32x8 acc[4][2];
#pragma unroll
  for (int i = 0; i < 4; i++)
#pragma unroll
    for (int j = 0; j < 2; j++) acc[i][j] = zf;

  int nk = K / 32;
  stage(0, 0);
  for (int kb = 0; kb < nk; kb++) {
    int cur = kb & 1;
    if (kb + 1 < nk) {
      stage(cur ^ 1, (kb + 1) * 32);                    // overlap next copy
      asm volatile("s_wait_asynccnt 4" ::: "memory");   // current buf resident
    } else {
      asm volatile("s_wait_asynccnt 0" ::: "memory");
    }
    __syncthreads();

    bf16x16 af[4], bfr[2];
#pragma unroll
    for (int i = 0; i < 4; i++) {
      const __bf16* p = &tA[cur][(wm * 64 + i * 16 + arow) * 32 + akh];
      af[i] = cat8(*(const bf16x8*)p, *(const bf16x8*)(p + 16));
    }
#pragma unroll
    for (int j = 0; j < 2; j++) {
      const __bf16* p = &tB[cur][(wn * 32 + j * 16 + arow) * 32 + bkh];
      bfr[j] = cat8(*(const bf16x8*)p, *(const bf16x8*)(p + 16));
    }
#pragma unroll
    for (int i = 0; i < 4; i++)
#pragma unroll
      for (int j = 0; j < 2; j++) acc[i][j] = wmma_bf16(af[i], bfr[j], acc[i][j]);

    asm volatile("s_wait_dscnt 0" ::: "memory");  // LDS reads done before reuse
    __syncthreads();
  }

  int rlo = (lane < 16) ? 0 : 8, ccol = lane & 15;
#pragma unroll
  for (int i = 0; i < 4; i++)
#pragma unroll
    for (int j = 0; j < 2; j++)
#pragma unroll
      for (int r = 0; r < 8; r++)
        C[(size_t)(rowBase + wm * 64 + i * 16 + rlo + r) * N +
          (colBase + wn * 32 + j * 16 + ccol)] = acc[i][j][r];
}

// ------------------------------------------------- RMSNorm + RoPE + layouts
// One block per (b,t,task): task 0..15 = q heads, 16..23 = k heads, 24..31 = v.
// Writes: QB [b,hq,t,d] bf16 (q scaled by 1/16), KB [b,hkv,t,d] bf16,
//         VT [b,hkv,d,t] bf16 (transposed -> Bt layout for P*V).
__global__ __launch_bounds__(256) void k_postprocess(
    const float* __restrict__ qf, const float* __restrict__ kf,
    const float* __restrict__ vf, const float* __restrict__ ksc,
    const float* __restrict__ vsc, const int* __restrict__ pos,
    __bf16* __restrict__ qb, __bf16* __restrict__ kb, __bf16* __restrict__ vt) {
  int blk  = blockIdx.x;
  int task = blk & 31;
  int bt   = blk >> 5;            // b*S + t
  int t    = bt & (SS - 1);
  int b    = bt >> 11;
  int d    = threadIdx.x;

  __shared__ float red[256];
  __shared__ float sv[256];

  bool isq = task < 16;
  bool isv = task >= 24;
  int h;
  float val;
  if (isq)      { h = task;      val = qf[(size_t)bt * DDn  + h * HDn + d]; }
  else if (!isv){ h = task - 16; val = kf[(size_t)bt * DKVn + h * HDn + d]; }
  else          { h = task - 24; val = vf[(size_t)bt * DKVn + h * HDn + d]; }

  if (!isq) {  // Gemma RMSNorm over head_dim with (1 + scale)
    red[d] = val * val;
    __syncthreads();
    for (int s2 = 128; s2 > 0; s2 >>= 1) {
      if (d < s2) red[d] += red[d + s2];
      __syncthreads();
    }
    float inv = rsqrtf(red[0] * (1.0f / HDn) + 1e-6f);
    const float* sc = isv ? vsc : ksc;
    val = val * inv * (1.0f + sc[d]);
  }

  if (isv) {  // no rope on v; store transposed
    vt[((size_t)(b * HKVn + h) * HDn + d) * SS + t] = f2bf(val);
    return;   // uniform per block, no divergent barrier
  }

  // RoPE (q and k)
  sv[d] = val;
  __syncthreads();
  int   i      = d & 127;
  float first  = sv[i];
  float second = sv[i + 128];
  float ts     = __powf(10000.0f, (float)i * (1.0f / 128.0f));
  float sinu   = (float)pos[(size_t)b * SS + t] / ts;
  float sn, cs;
  __sincosf(sinu, &sn, &cs);
  float out = (d < 128) ? (first * cs - second * sn) : (second * cs + first * sn);

  if (isq) qb[((size_t)(b * HQn  + h) * SS + t) * HDn + d] = f2bf(out * 0.0625f);
  else     kb[((size_t)(b * HKVn + h) * SS + t) * HDn + d] = f2bf(out);
}

// ------------------------------------------------------ flash attention
// Block = 8 waves; each wave owns 16 query rows (128 rows per block).
// Per causal 32-key block: K/V tiles async-copied to LDS once (kills the 8x
// per-wave duplication), 16 WMMA (scores, K=256) + online softmax (xor
// shuffles in 16-lane halves) + LDS C->A relayout + 16 WMMA (P*V).
__global__ __launch_bounds__(256) void k_attn(const __bf16* __restrict__ qb,
                                              const __bf16* __restrict__ kb,
                                              const __bf16* __restrict__ vt,
                                              __bf16* __restrict__ attb) {
  int tid  = threadIdx.x;
  int lane = tid & 31;
  int wave = tid >> 5;
  int qblk = blockIdx.x;          // 0..15
  int h    = blockIdx.y;          // 0..15
  int b    = blockIdx.z;          // 0..1
  int hk   = h >> 1;              // GQA group = 2
  int q0   = qblk * 128 + wave * 16;

  const int arow = lane & 15;
  const int akh  = (lane < 16) ? 0 : 8;
  const int bkh  = (lane < 16) ? 0 : 8;
  const int rlo  = (lane < 16) ? 0 : 8;

  const __bf16* qbase = qb + ((size_t)(b * HQn + h) * SS + q0) * HDn;
  const __bf16* kbase = kb + (size_t)(b * HKVn + hk) * SS * HDn;
  const __bf16* vbase = vt + (size_t)(b * HKVn + hk) * HDn * SS;

  __shared__ __align__(16) __bf16 tK[32 * 256];     // 16KB: K rows x d
  __shared__ __align__(16) __bf16 tV[256 * 32];     // 16KB: d rows x keys
  __shared__ __align__(16) __bf16 psh[8][16 * 32];  // 8KB: per-wave P staging

  // Preload the 16x256 Q tile as 8 A-fragments (stays in VGPRs all loop).
  bf16x16 qfr[8];
#pragma unroll
  for (int c = 0; c < 8; c++) {
    const __bf16* p = qbase + (size_t)arow * HDn + c * 32 + akh;
    qfr[c] = cat8(*(const bf16x8*)p, *(const bf16x8*)(p + 16));
  }

  const f32x8 zf = {};
  f32x8 acc[16];
#pragma unroll
  for (int j = 0; j < 16; j++) acc[j] = zf;

  float m[8], l[8];
#pragma unroll
  for (int r = 0; r < 8; r++) { m[r] = -1e30f; l[r] = 0.0f; }

  __bf16* pw = &psh[wave][0];

  int nkb = qblk * 4 + 4;  // causal: keys up to (q0_block + 128)
  for (int kbi = 0; kbi < nkb; kbi++) {
    int kc0 = kbi * 32;

    // Async stage K tile (32x256) and V^T tile (256x32) into LDS.
    // 256 threads x 4 iters x 2 ops = 8 async ops/thread.
#pragma unroll
    for (int i = 0; i < 4; i++) {
      int idx  = i * 256 + tid;       // 0..1023
      int krow = idx >> 5, kch = idx & 31;   // 32 rows x 32 chunks of 16B
      async_cp16(kbase + (size_t)(kc0 + krow) * HDn + kch * 8,
                 &tK[krow * 256 + kch * 8]);
      int vrow = idx >> 2, vch = idx & 3;    // 256 rows x 4 chunks of 16B
      async_cp16(vbase + (size_t)vrow * SS + kc0 + vch * 8,
                 &tV[vrow * 32 + vch * 8]);
    }
    asm volatile("s_wait_asynccnt 0" ::: "memory");
    __syncthreads();

    // scores: 16 rows x 32 keys, K = 256 (B-fragments from LDS)
    f32x8 sA = zf, sB = zf;
#pragma unroll
    for (int c = 0; c < 8; c++) {
      const __bf16* pkA = &tK[arow * 256 + c * 32 + bkh];
      const __bf16* pkB = &tK[(16 + arow) * 256 + c * 32 + bkh];
      bf16x16 bA = cat8(*(const bf16x8*)pkA, *(const bf16x8*)(pkA + 16));
      bf16x16 bB = cat8(*(const bf16x8*)pkB, *(const bf16x8*)(pkB + 16));
      sA = wmma_bf16(qfr[c], bA, sA);
      sB = wmma_bf16(qfr[c], bB, sB);
    }

    // causal mask (analytic tril)
    int kA = kc0 + arow;
#pragma unroll
    for (int r = 0; r < 8; r++) {
      int qrow = q0 + rlo + r;
      if (kA > qrow)      sA[r] = -1e10f;
      if (kA + 16 > qrow) sB[r] = -1e10f;
    }

    // online softmax: row max / row sum via xor shuffles inside 16-lane halves
    float rowm[8];
#pragma unroll
    for (int r = 0; r < 8; r++) rowm[r] = fmaxf(sA[r], sB[r]);
#pragma unroll
    for (int off = 1; off < 16; off <<= 1)
#pragma unroll
      for (int r = 0; r < 8; r++)
        rowm[r] = fmaxf(rowm[r], __shfl_xor(rowm[r], off, 32));

    float scl[8];
#pragma unroll
    for (int r = 0; r < 8; r++) {
      float mn = fmaxf(m[r], rowm[r]);
      scl[r]   = __expf(m[r] - mn);
      m[r]     = mn;
    }
    float rs[8];
#pragma unroll
    for (int r = 0; r < 8; r++) {
      sA[r] = __expf(sA[r] - m[r]);
      sB[r] = __expf(sB[r] - m[r]);
      rs[r] = sA[r] + sB[r];
    }
#pragma unroll
    for (int off = 1; off < 16; off <<= 1)
#pragma unroll
      for (int r = 0; r < 8; r++) rs[r] += __shfl_xor(rs[r], off, 32);
#pragma unroll
    for (int r = 0; r < 8; r++) l[r] = l[r] * scl[r] + rs[r];

    // rescale output accumulators
#pragma unroll
    for (int j = 0; j < 16; j++)
#pragma unroll
      for (int r = 0; r < 8; r++) acc[j][r] *= scl[r];

    // P (C-layout) -> LDS -> A-fragment layout
#pragma unroll
    for (int r = 0; r < 8; r++) {
      pw[(rlo + r) * 32 + arow]      = f2bf(sA[r]);
      pw[(rlo + r) * 32 + 16 + arow] = f2bf(sB[r]);
    }
    const __bf16* pr = pw + arow * 32 + akh;
    bf16x16 pf = cat8(*(const bf16x8*)pr, *(const bf16x8*)(pr + 16));

    // P(16x32) @ V(32x256): V^T tile in LDS is the Bt layout
#pragma unroll
    for (int j = 0; j < 16; j++) {
      const __bf16* pv = &tV[(j * 16 + arow) * 32 + bkh];
      bf16x16 vfr = cat8(*(const bf16x8*)pv, *(const bf16x8*)(pv + 16));
      acc[j] = wmma_bf16(pf, vfr, acc[j]);
    }

    asm volatile("s_wait_dscnt 0" ::: "memory");  // reads done before restage
    __syncthreads();
  }

  // epilogue: normalize, write [b, t, h, d] as bf16 for the Wo GEMM
  float li[8];
#pragma unroll
  for (int r = 0; r < 8; r++) li[r] = 1.0f / l[r];
  __bf16* ob = attb + (size_t)b * SS * DDn;
#pragma unroll
  for (int j = 0; j < 16; j++)
#pragma unroll
    for (int r = 0; r < 8; r++) {
      int trow = q0 + rlo + r;
      ob[(size_t)trow * DDn + h * HDn + j * 16 + arow] = f2bf(acc[j][r] * li[r]);
    }
}

// ---------------------------------------------------------------- launcher
extern "C" void kernel_launch(void* const* d_in, const int* in_sizes, int n_in,
                              void* d_out, int out_size, void* d_ws, size_t ws_size,
                              hipStream_t stream) {
  const float* x   = (const float*)d_in[0];
  const float* Wq  = (const float*)d_in[1];
  const float* Wk  = (const float*)d_in[2];
  const float* Wv  = (const float*)d_in[3];
  const float* Wo  = (const float*)d_in[4];
  const float* ksc = (const float*)d_in[5];
  const float* vsc = (const float*)d_in[6];
  // d_in[7] = mask: fixed causal tril, recomputed analytically in-kernel.
  const int* pos = (const int*)d_in[8];
  (void)in_sizes; (void)n_in; (void)out_size; (void)ws_size;

  char*  ws  = (char*)d_ws;
  size_t off = 0;
  auto alloc = [&](size_t bytes) { char* p = ws + off; off += bytes; return p; };

  __bf16* XB  = (__bf16*)alloc((size_t)BSn * DDn * 2);
  __bf16* WQT = (__bf16*)alloc((size_t)DDn * DDn * 2);
  __bf16* WKT = (__bf16*)alloc((size_t)DKVn * DDn * 2);
  __bf16* WVT = (__bf16*)alloc((size_t)DKVn * DDn * 2);
  __bf16* WOT = (__bf16*)alloc((size_t)DDn * DDn * 2);
  float*  QF  = (float*)alloc((size_t)BSn * DDn * 4);
  float*  KF  = (float*)alloc((size_t)BSn * DKVn * 4);
  float*  VF  = (float*)alloc((size_t)BSn * DKVn * 4);
  __bf16* QB  = (__bf16*)alloc((size_t)BSn * DDn * 2);
  __bf16* KB  = (__bf16*)alloc((size_t)BSn * DKVn * 2);
  __bf16* VT  = (__bf16*)alloc((size_t)BSn * DKVn * 2);
  __bf16* ATTB = (__bf16*)QF;  // alias: QF is dead after k_postprocess

  // 1) precision conversion + weight transposes (Bt layouts)
  k_cvt_bf16<<<(BSn * DDn + 255) / 256, 256, 0, stream>>>(x, XB, BSn * DDn);
  k_transpose_bf16<<<(DDn * DDn + 255) / 256, 256, 0, stream>>>(Wq, WQT, DDn, DDn);
  k_transpose_bf16<<<(DKVn * DDn + 255) / 256, 256, 0, stream>>>(Wk, WKT, DKVn, DDn);
  k_transpose_bf16<<<(DKVn * DDn + 255) / 256, 256, 0, stream>>>(Wv, WVT, DKVn, DDn);
  k_transpose_bf16<<<(DDn * DDn + 255) / 256, 256, 0, stream>>>(Wo, WOT, DDn, DDn);

  // 2) QKV projections (WMMA, async-LDS staged)
  k_gemm_bf16<<<dim3(DDn / 128, BSn / 128), 256, 0, stream>>>(XB, WQT, QF, BSn, DDn, DDn);
  k_gemm_bf16<<<dim3(DKVn / 128, BSn / 128), 256, 0, stream>>>(XB, WKT, KF, BSn, DKVn, DDn);
  k_gemm_bf16<<<dim3(DKVn / 128, BSn / 128), 256, 0, stream>>>(XB, WVT, VF, BSn, DKVn, DDn);

  // 3) RMSNorm + RoPE + attention-friendly layouts
  k_postprocess<<<BSn * 32, 256, 0, stream>>>(QF, KF, VF, ksc, vsc, pos, QB, KB, VT);

  // 4) causal flash attention (WMMA, async-LDS staged K/V)
  k_attn<<<dim3(SS / 128, HQn, BB), 256, 0, stream>>>(QB, KB, VT, ATTB);

  // 5) output projection (WMMA) -> fp32 d_out [B,S,D]
  k_gemm_bf16<<<dim3(DDn / 128, BSn / 128), 256, 0, stream>>>(ATTB, WOT, (float*)d_out,
                                                              BSn, DDn, DDn);
}
